// EquivariantGridModel_8864812499083
// MI455X (gfx1250) — matrix operand order repaired and checked by reading, not compiled, not run on hardware
//
#include <hip/hip_runtime.h>
#include <hip/hip_bf16.h>

// ---------------------------------------------------------------------------
// EquivariantGridModel for MI455X (gfx1250), wave32 + WMMA f16 16x16x32.
// Dead-code-eliminated: the equivariant branch (edge_index, eq.net1/net2)
// never reaches the output (reference `del orientations`).
// ---------------------------------------------------------------------------

typedef __attribute__((ext_vector_type(16))) _Float16 v16h;
typedef __attribute__((ext_vector_type(8)))  _Float16 v8h;
typedef __attribute__((ext_vector_type(8)))  float    v8f;

#define DEVI __device__ __forceinline__

static constexpr int NN  = 32768;    // atoms
static constexpr int EE  = 524288;   // atom-atom edges (unused)
static constexpr int EG  = 524288;   // atom-grid edges
static constexpr int BB  = 64;       // batch
static constexpr int GG  = 46656;    // grid points (64*9^3)
static constexpr int HH  = 128;

// ---------------- WMMA tile loaders (CDNA5 ISA 7.12.2 layouts) -------------

// A-matrix 16x32 f16: lane l -> row l&15; K chunks at (l>>4)*8 and +16.
DEVI v16h ldA(const _Float16* p, int ld) {
  int l = threadIdx.x & 31;
  const _Float16* r = p + (size_t)(l & 15) * ld + ((l >> 4) << 3);
  v16h o;
#pragma unroll
  for (int i = 0; i < 8; ++i) { o[i] = r[i]; o[8 + i] = r[16 + i]; }
  return o;
}

// B-matrix 32x16 f16 loaded from W^T row-major [N][K]:
// lane l -> col l&15; K chunks at (l>>4)*8 and +16.
DEVI v16h ldB(const _Float16* p, int ld) {
  int l = threadIdx.x & 31;
  const _Float16* r = p + (size_t)(l & 15) * ld + ((l >> 4) << 3);
  v16h o;
#pragma unroll
  for (int i = 0; i < 8; ++i) { o[i] = r[i]; o[8 + i] = r[16 + i]; }
  return o;
}

DEVI v8f wmma16(v16h a, v16h b, v8f c) {
  return __builtin_amdgcn_wmma_f32_16x16x32_f16(false, a, false, b, (short)0, c,
                                                false, false);
}

DEVI v8f bsplat(float v) { v8f a = {v, v, v, v, v, v, v, v}; return a; }
DEVI float silu(float v) { return v / (1.0f + __expf(-v)); }

// ---------------- weight prep -------------------------------------------------

// wt[n*Kpad + k] = (k < K) ? w[k*Nout + n] : 0   (W^T, f16, K zero-padded)
__global__ void k_wt(const float* __restrict__ w, _Float16* __restrict__ wt,
                     int K, int Nout, int Kpad) {
  int i = blockIdx.x * blockDim.x + threadIdx.x;
  if (i >= Nout * Kpad) return;
  int n = i / Kpad, k = i % Kpad;
  wt[i] = (k < K) ? (_Float16)w[(size_t)k * Nout + n] : (_Float16)0.0f;
}

// conv weights: wt[o][co][ci] = w[co][ci][o]  (f16)
__global__ void k_wconv(const float* __restrict__ w, _Float16* __restrict__ wt,
                        int cout, int cin, int ks3) {
  int i = blockIdx.x * blockDim.x + threadIdx.x;
  if (i >= cout * cin * ks3) return;
  int o = i / (cout * cin);
  int r = i % (cout * cin);
  int co = r / cin, ci = r % cin;
  wt[i] = (_Float16)w[((size_t)co * cin + ci) * ks3 + o];
}

// ---------------- node MLP: nf = MLP(embed[atoms]) -> f16 [N][128] -----------

__global__ void __launch_bounds__(128) k_node(
    const int* __restrict__ atoms, const float* __restrict__ embed,
    const _Float16* __restrict__ w1, const float* __restrict__ b1,
    const _Float16* __restrict__ w2, const float* __restrict__ b2,
    _Float16* __restrict__ nf16) {
  __shared__ _Float16 stage[4][16 * 136];
  int wave = threadIdx.x >> 5, l = threadIdx.x & 31;
  int m = l & 15, half = l >> 4;
  int row0 = (blockIdx.x * 4 + wave) * 16;
  const float* arow = embed + (size_t)atoms[row0 + m] * HH;
  v16h A[4];
#pragma unroll
  for (int ks = 0; ks < 4; ++ks) {
    int kb = ks * 32 + half * 8;
    v16h a;
#pragma unroll
    for (int i = 0; i < 8; ++i) {
      a[i] = (_Float16)arow[kb + i];
      a[8 + i] = (_Float16)arow[kb + 16 + i];
    }
    A[ks] = a;
  }
  _Float16* st = stage[wave];
#pragma unroll
  for (int nt = 0; nt < 8; ++nt) {
    v8f acc = bsplat(b1[nt * 16 + m]);
#pragma unroll
    for (int ks = 0; ks < 4; ++ks)
      acc = wmma16(A[ks], ldB(w1 + (size_t)nt * 16 * 128 + ks * 32, 128), acc);
#pragma unroll
    for (int v = 0; v < 8; ++v)
      st[(v + half * 8) * 136 + nt * 16 + m] = (_Float16)silu(acc[v]);
  }
  v16h A2[4];
#pragma unroll
  for (int ks = 0; ks < 4; ++ks) A2[ks] = ldA(st + ks * 32, 136);
#pragma unroll
  for (int nt = 0; nt < 8; ++nt) {
    v8f acc = bsplat(b2[nt * 16 + m]);
#pragma unroll
    for (int ks = 0; ks < 4; ++ks)
      acc = wmma16(A2[ks], ldB(w2 + (size_t)nt * 16 * 128 + ks * 32, 128), acc);
#pragma unroll
    for (int v = 0; v < 8; ++v)
      nf16[(size_t)(row0 + v + half * 8) * HH + nt * 16 + m] = (_Float16)acc[v];
  }
}

// ---------------- fused edge+msg MLPs + scatter segment-sum ------------------

__global__ void __launch_bounds__(128) k_edge(
    const int* __restrict__ src, const int* __restrict__ tgt,
    const float* __restrict__ x, const float* __restrict__ gc,
    const _Float16* __restrict__ we1, const float* __restrict__ be1,
    const _Float16* __restrict__ we2, const float* __restrict__ be2,
    const _Float16* __restrict__ wm1, const float* __restrict__ bm1,
    const _Float16* __restrict__ wm2, const float* __restrict__ bm2,
    const _Float16* __restrict__ nf16, float* __restrict__ gf,
    float* __restrict__ cnt) {
  __shared__ _Float16 xa[4][16 * 264];   // [16 rows][nf(128) | ef(128) | pad]
  __shared__ _Float16 hb[4][16 * 136];   // hidden staging
  int wave = threadIdx.x >> 5, l = threadIdx.x & 31;
  int m = l & 15, half = l >> 4;
  int e0 = (blockIdx.x * 4 + wave) * 16;
  int s = src[e0 + m], t = tgt[e0 + m];

  // gather nf[src] (f16) into xa[:,0:128]
  {
    const v8h* nfr = (const v8h*)(nf16 + (size_t)s * HH + half * 64);
    v8h* dst = (v8h*)(xa[wave] + m * 264 + half * 64);
#pragma unroll
    for (int j = 0; j < 8; ++j) dst[j] = nfr[j];
  }
  // geometry A-tile, K=6 (zero-padded to 32)
  v16h a1;
#pragma unroll
  for (int i = 0; i < 16; ++i) a1[i] = (_Float16)0.0f;
  if (half == 0) {
    a1[0] = (_Float16)x[s * 3 + 0]; a1[1] = (_Float16)x[s * 3 + 1];
    a1[2] = (_Float16)x[s * 3 + 2];
    a1[3] = (_Float16)gc[t * 3 + 0]; a1[4] = (_Float16)gc[t * 3 + 1];
    a1[5] = (_Float16)gc[t * 3 + 2];
  }
  _Float16* hrow = hb[wave];
  // edge MLP layer 1
#pragma unroll
  for (int nt = 0; nt < 8; ++nt) {
    v8f acc = bsplat(be1[nt * 16 + m]);
    acc = wmma16(a1, ldB(we1 + (size_t)nt * 16 * 32, 32), acc);
#pragma unroll
    for (int v = 0; v < 8; ++v)
      hrow[(v + half * 8) * 136 + nt * 16 + m] = (_Float16)silu(acc[v]);
  }
  // edge MLP layer 2 -> ef into xa[:,128:256]
  {
    v16h A2[4];
#pragma unroll
    for (int ks = 0; ks < 4; ++ks) A2[ks] = ldA(hrow + ks * 32, 136);
#pragma unroll
    for (int nt = 0; nt < 8; ++nt) {
      v8f acc = bsplat(be2[nt * 16 + m]);
#pragma unroll
      for (int ks = 0; ks < 4; ++ks)
        acc = wmma16(A2[ks], ldB(we2 + (size_t)nt * 16 * 128 + ks * 32, 128), acc);
#pragma unroll
      for (int v = 0; v < 8; ++v)
        xa[wave][(v + half * 8) * 264 + 128 + nt * 16 + m] = (_Float16)acc[v];
    }
  }
  // msg MLP layer 1 (K = 256 : [nf || ef])
  v16h AM[8];
#pragma unroll
  for (int ks = 0; ks < 8; ++ks) AM[ks] = ldA(xa[wave] + ks * 32, 264);
#pragma unroll
  for (int nt = 0; nt < 8; ++nt) {
    v8f acc = bsplat(bm1[nt * 16 + m]);
#pragma unroll
    for (int ks = 0; ks < 8; ++ks)
      acc = wmma16(AM[ks], ldB(wm1 + (size_t)nt * 16 * 256 + ks * 32, 256), acc);
#pragma unroll
    for (int v = 0; v < 8; ++v)
      hrow[(v + half * 8) * 136 + nt * 16 + m] = (_Float16)silu(acc[v]);
  }
  // msg MLP layer 2 + atomic scatter into grid accumulator
  v16h A3[4];
#pragma unroll
  for (int ks = 0; ks < 4; ++ks) A3[ks] = ldA(hrow + ks * 32, 136);
  int trow[8];
#pragma unroll
  for (int v = 0; v < 8; ++v) trow[v] = tgt[e0 + v + half * 8];
#pragma unroll
  for (int nt = 0; nt < 8; ++nt) {
    v8f acc = bsplat(bm2[nt * 16 + m]);
#pragma unroll
    for (int ks = 0; ks < 4; ++ks)
      acc = wmma16(A3[ks], ldB(wm2 + (size_t)nt * 16 * 128 + ks * 32, 128), acc);
#pragma unroll
    for (int v = 0; v < 8; ++v)
      atomicAdd(&gf[(size_t)trow[v] * HH + nt * 16 + m], acc[v]);
  }
  if (m == 0) {  // lanes 0 (rows 0-7) and 16 (rows 8-15): one count per edge
#pragma unroll
    for (int v = 0; v < 8; ++v) atomicAdd(&cnt[trow[v]], 1.0f);
  }
}

// ---------------- upd MLP on segment mean -> CNN input (channels-last) -------

__global__ void __launch_bounds__(128) k_upd(
    const float* __restrict__ gf, const float* __restrict__ cnt,
    const _Float16* __restrict__ w1, const float* __restrict__ b1,
    const _Float16* __restrict__ w2, const float* __restrict__ b2,
    float* __restrict__ t0) {
  __shared__ _Float16 stage[4][16 * 136];
  int wave = threadIdx.x >> 5, l = threadIdx.x & 31;
  int m = l & 15, half = l >> 4;
  int row0 = (blockIdx.x * 4 + wave) * 16;
  const float* grow = gf + (size_t)(row0 + m) * HH;
  float sc = 1.0f / fmaxf(cnt[row0 + m], 1.0f);
  v16h A[4];
#pragma unroll
  for (int ks = 0; ks < 4; ++ks) {
    int kb = ks * 32 + half * 8;
    v16h a;
#pragma unroll
    for (int i = 0; i < 8; ++i) {
      a[i] = (_Float16)(grow[kb + i] * sc);
      a[8 + i] = (_Float16)(grow[kb + 16 + i] * sc);
    }
    A[ks] = a;
  }
  _Float16* st = stage[wave];
#pragma unroll
  for (int nt = 0; nt < 8; ++nt) {
    v8f acc = bsplat(b1[nt * 16 + m]);
#pragma unroll
    for (int ks = 0; ks < 4; ++ks)
      acc = wmma16(A[ks], ldB(w1 + (size_t)nt * 16 * 128 + ks * 32, 128), acc);
#pragma unroll
    for (int v = 0; v < 8; ++v)
      st[(v + half * 8) * 136 + nt * 16 + m] = (_Float16)silu(acc[v]);
  }
  v16h A2[4];
#pragma unroll
  for (int ks = 0; ks < 4; ++ks) A2[ks] = ldA(st + ks * 32, 136);
#pragma unroll
  for (int nt = 0; nt < 8; ++nt) {
    v8f acc = bsplat(b2[nt * 16 + m]);
#pragma unroll
    for (int ks = 0; ks < 4; ++ks)
      acc = wmma16(A2[ks], ldB(w2 + (size_t)nt * 16 * 128 + ks * 32, 128), acc);
#pragma unroll
    for (int v = 0; v < 8; ++v)
      t0[(size_t)(row0 + v + half * 8) * HH + nt * 16 + m] = acc[v];
  }
}

// ---------------- implicit-GEMM 3D conv (channels-last), WMMA ----------------

template <int CIN, int COUT, int KS>
__global__ void __launch_bounds__(128) k_conv(
    const float* __restrict__ x, float* __restrict__ y,
    const _Float16* __restrict__ wt, int Di, int Do, int stride) {
  constexpr int LDP = CIN + 8;
  constexpr int NT = COUT / 16;
  constexpr int NK = CIN / 32;
  constexpr int PAD = (KS == 3) ? 1 : 0;
  __shared__ _Float16 stage[4][16 * LDP];
  int wave = threadIdx.x >> 5, l = threadIdx.x & 31;
  int m = l & 15, half = l >> 4;
  int p0 = (blockIdx.x * 4 + wave) * 16;
  int p = p0 + m;
  int xo = p % Do, yo = (p / Do) % Do, zo = (p / (Do * Do)) % Do;
  int b = p / (Do * Do * Do);
  v8f acc[NT];
#pragma unroll
  for (int nt = 0; nt < NT; ++nt) acc[nt] = bsplat(0.0f);
  _Float16* st = stage[wave];
  for (int o = 0; o < KS * KS * KS; ++o) {
    int kz = o / (KS * KS), ky = (o / KS) % KS, kx = o % KS;
    int zi = zo * stride - PAD + kz;
    int yi = yo * stride - PAD + ky;
    int xi = xo * stride - PAD + kx;
    bool ok = (unsigned)zi < (unsigned)Di && (unsigned)yi < (unsigned)Di &&
              (unsigned)xi < (unsigned)Di;
    const float* xr =
        x + ((size_t)((b * Di + zi) * Di + yi) * Di + xi) * CIN + half * (CIN / 2);
    _Float16* sr = st + m * LDP + half * (CIN / 2);
#pragma unroll 8
    for (int j = 0; j < CIN / 2; ++j)
      sr[j] = ok ? (_Float16)xr[j] : (_Float16)0.0f;
#pragma unroll
    for (int ks = 0; ks < NK; ++ks) {
      v16h A = ldA(st + ks * 32, LDP);
      const _Float16* wb = wt + (size_t)o * COUT * CIN + ks * 32;
#pragma unroll
      for (int nt = 0; nt < NT; ++nt)
        acc[nt] = wmma16(A, ldB(wb + (size_t)nt * 16 * CIN, CIN), acc[nt]);
    }
  }
#pragma unroll
  for (int nt = 0; nt < NT; ++nt)
#pragma unroll
    for (int v = 0; v < 8; ++v)
      y[(size_t)(p0 + v + half * 8) * COUT + nt * 16 + m] = acc[nt][v];
}

// ---------------- training-mode BatchNorm helpers ---------------------------

__global__ void k_stats(const float* __restrict__ x, float* __restrict__ st,
                        int npos) {
  int C = blockDim.x, c = threadIdx.x;
  float s = 0.0f, ss = 0.0f;
  for (int p = blockIdx.x; p < npos; p += gridDim.x) {
    float v = x[(size_t)p * C + c];
    s += v; ss += v * v;
  }
  atomicAdd(&st[c], s);
  atomicAdd(&st[C + c], ss);
}

__global__ void k_bn_apply(const float* __restrict__ x, float* __restrict__ y,
                           const float* __restrict__ st,
                           const float* __restrict__ g,
                           const float* __restrict__ bb, int total, int C,
                           float inv_n, int do_silu) {
  int i = blockIdx.x * blockDim.x + threadIdx.x;
  if (i >= total) return;
  int c = i % C;
  float mean = st[c] * inv_n;
  float var = st[C + c] * inv_n - mean * mean;
  float sc = g[c] * rsqrtf(var + 1e-5f);
  float v = (x[i] - mean) * sc + bb[c];
  y[i] = do_silu ? silu(v) : v;
}

__global__ void k_bn2_combine(const float* __restrict__ ym,
                              const float* __restrict__ sm,
                              const float* __restrict__ sty,
                              const float* __restrict__ sts,
                              const float* __restrict__ g2,
                              const float* __restrict__ b2,
                              const float* __restrict__ gs,
                              const float* __restrict__ bs,
                              float* __restrict__ out, int total, int C,
                              float inv_n) {
  int i = blockIdx.x * blockDim.x + threadIdx.x;
  if (i >= total) return;
  int c = i % C;
  float my = sty[c] * inv_n;
  float vy = sty[C + c] * inv_n - my * my;
  float a = (ym[i] - my) * (g2[c] * rsqrtf(vy + 1e-5f)) + b2[c];
  float ms = sts[c] * inv_n;
  float vs = sts[C + c] * inv_n - ms * ms;
  float s = (sm[i] - ms) * (gs[c] * rsqrtf(vs + 1e-5f)) + bs[c];
  out[i] = silu(a + s);
}

// ---------------- global average pool + FC ----------------------------------

__global__ void k_pool_fc(const float* __restrict__ xin,
                          const float* __restrict__ fcw,
                          const float* __restrict__ fcb,
                          float* __restrict__ out) {
  __shared__ float pooled[256];
  int b = blockIdx.x, c = threadIdx.x;
  float s = 0.0f;
  for (int p = 0; p < 27; ++p) s += xin[((size_t)b * 27 + p) * 256 + c];
  pooled[c] = s * (1.0f / 27.0f);
  __syncthreads();
  if (c < 20) {
    float acc = fcb[c];
    for (int k = 0; k < 256; ++k) acc += pooled[k] * fcw[k * 20 + c];
    out[b * 20 + c] = acc;
  }
}

// ---------------- host-side residual block driver ---------------------------

template <int CIN, int COUT>
static void run_block(const float* in, float* a, float* bbuf, float* out,
                      const float* const* P, const _Float16* w1t,
                      const _Float16* w2t, const _Float16* wst, float* st,
                      int Di, int stride, hipStream_t stream) {
  int Do = (Di + 2 - 3) / stride + 1;
  int Pn = BB * Do * Do * Do;
  int blocks = Pn / 64;
  k_conv<CIN, COUT, 3><<<blocks, 128, 0, stream>>>(in, a, w1t, Di, Do, stride);
  k_stats<<<64, COUT, 0, stream>>>(a, st + 0, Pn);
  k_bn_apply<<<(Pn * COUT + 255) / 256, 256, 0, stream>>>(
      a, a, st + 0, P[1], P[2], Pn * COUT, COUT, 1.0f / Pn, 1);
  k_conv<COUT, COUT, 3><<<blocks, 128, 0, stream>>>(a, bbuf, w2t, Do, Do, 1);
  k_stats<<<64, COUT, 0, stream>>>(bbuf, st + 512, Pn);
  k_conv<CIN, COUT, 1><<<blocks, 128, 0, stream>>>(in, a, wst, Di, Do, stride);
  k_stats<<<64, COUT, 0, stream>>>(a, st + 1024, Pn);
  k_bn2_combine<<<(Pn * COUT + 255) / 256, 256, 0, stream>>>(
      bbuf, a, st + 512, st + 1024, P[4], P[5], P[7], P[8], out, Pn * COUT,
      COUT, 1.0f / Pn);
}

// ---------------- launch ----------------------------------------------------

extern "C" void kernel_launch(void* const* d_in, const int* in_sizes, int n_in,
                              void* d_out, int out_size, void* d_ws,
                              size_t ws_size, hipStream_t stream) {
  (void)in_sizes; (void)n_in; (void)out_size; (void)ws_size;
  // input order (setup_inputs dict order, params flattened leaf-by-leaf)
  const int*   atoms = (const int*)d_in[0];
  const float* x     = (const float*)d_in[1];
  // d_in[2] edge_index: dead (equivariant branch never reaches output)
  const float* gcoords = (const float*)d_in[3];
  const int*   gedge   = (const int*)d_in[4];
  const int*   src = gedge;
  const int*   tgt = gedge + EG;
  const float* embed = (const float*)d_in[6];
  // eq.* = d_in[7..14]: dead
  const float* nd_w1 = (const float*)d_in[15]; const float* nd_b1 = (const float*)d_in[16];
  const float* nd_w2 = (const float*)d_in[17]; const float* nd_b2 = (const float*)d_in[18];
  const float* ed_w1 = (const float*)d_in[19]; const float* ed_b1 = (const float*)d_in[20];
  const float* ed_w2 = (const float*)d_in[21]; const float* ed_b2 = (const float*)d_in[22];
  const float* ms_w1 = (const float*)d_in[23]; const float* ms_b1 = (const float*)d_in[24];
  const float* ms_w2 = (const float*)d_in[25]; const float* ms_b2 = (const float*)d_in[26];
  const float* up_w1 = (const float*)d_in[27]; const float* up_b1 = (const float*)d_in[28];
  const float* up_w2 = (const float*)d_in[29]; const float* up_b2 = (const float*)d_in[30];
  const float* cnn_g = (const float*)d_in[31]; const float* cnn_b = (const float*)d_in[32];
  const float* fc_w = (const float*)d_in[87];  const float* fc_b = (const float*)d_in[88];

  // workspace bump allocator
  char* base = (char*)d_ws;
  size_t off = 0;
  auto alloc = [&](size_t bytes) -> void* {
    void* p = base + off;
    off = (off + bytes + 255) & ~(size_t)255;
    return p;
  };
  _Float16* WT_ND1 = (_Float16*)alloc(128 * 128 * 2);
  _Float16* WT_ND2 = (_Float16*)alloc(128 * 128 * 2);
  _Float16* WT_ED1 = (_Float16*)alloc(128 * 32 * 2);
  _Float16* WT_ED2 = (_Float16*)alloc(128 * 128 * 2);
  _Float16* WT_MS1 = (_Float16*)alloc(128 * 256 * 2);
  _Float16* WT_MS2 = (_Float16*)alloc(128 * 128 * 2);
  _Float16* WT_UP1 = (_Float16*)alloc(128 * 128 * 2);
  _Float16* WT_UP2 = (_Float16*)alloc(128 * 128 * 2);
  _Float16* NF16 = (_Float16*)alloc((size_t)NN * HH * 2);
  float* GF  = (float*)alloc((size_t)GG * HH * 4);
  float* CNT = (float*)alloc((size_t)GG * 4);
  float* T0  = (float*)alloc((size_t)GG * HH * 4);
  float* ST  = (float*)alloc(19 * 512 * 4);  // 19 BN stat slots (sum|sumsq)
  size_t bufElems = (size_t)GG * 64;         // largest conv activation
  float* BUF1 = (float*)alloc(bufElems * 4);
  float* BUF2 = (float*)alloc(bufElems * 4);
  float* BUF3 = (float*)alloc(bufElems * 4);

  // CNN block weights (f16, [27][cout][cin] / [cout][cin])
  const int bin[6]  = {33, 42, 51, 60, 69, 78};
  const int cins[6] = {128, 64, 64, 128, 128, 256};
  const int cout6[6] = {64, 64, 128, 128, 256, 256};
  const float* BP[6][9];
  _Float16 *W1T[6], *W2T[6], *WST[6];
  for (int i = 0; i < 6; ++i) {
    for (int j = 0; j < 9; ++j) BP[i][j] = (const float*)d_in[bin[i] + j];
    W1T[i] = (_Float16*)alloc((size_t)27 * cout6[i] * cins[i] * 2);
    W2T[i] = (_Float16*)alloc((size_t)27 * cout6[i] * cout6[i] * 2);
    WST[i] = (_Float16*)alloc((size_t)cout6[i] * cins[i] * 2);
  }

  // ---- weight prep ----
  auto wt = [&](const float* w, _Float16* wtp, int K, int Nout, int Kpad) {
    int n = Nout * Kpad;
    k_wt<<<(n + 255) / 256, 256, 0, stream>>>(w, wtp, K, Nout, Kpad);
  };
  wt(nd_w1, WT_ND1, 128, 128, 128);
  wt(nd_w2, WT_ND2, 128, 128, 128);
  wt(ed_w1, WT_ED1, 6, 128, 32);
  wt(ed_w2, WT_ED2, 128, 128, 128);
  wt(ms_w1, WT_MS1, 256, 128, 256);
  wt(ms_w2, WT_MS2, 128, 128, 128);
  wt(up_w1, WT_UP1, 128, 128, 128);
  wt(up_w2, WT_UP2, 128, 128, 128);
  for (int i = 0; i < 6; ++i) {
    int n1 = 27 * cout6[i] * cins[i];
    int n2 = 27 * cout6[i] * cout6[i];
    int n3 = cout6[i] * cins[i];
    k_wconv<<<(n1 + 255) / 256, 256, 0, stream>>>(BP[i][0], W1T[i], cout6[i], cins[i], 27);
    k_wconv<<<(n2 + 255) / 256, 256, 0, stream>>>(BP[i][3], W2T[i], cout6[i], cout6[i], 27);
    k_wconv<<<(n3 + 255) / 256, 256, 0, stream>>>(BP[i][6], WST[i], cout6[i], cins[i], 1);
  }

  // ---- zero accumulators ----
  hipMemsetAsync(GF, 0, (size_t)GG * HH * 4, stream);
  hipMemsetAsync(CNT, 0, (size_t)GG * 4, stream);
  hipMemsetAsync(ST, 0, 19 * 512 * 4, stream);

  // ---- graph pipeline ----
  k_node<<<NN / 64, 128, 0, stream>>>(atoms, embed, WT_ND1, nd_b1, WT_ND2, nd_b2, NF16);
  k_edge<<<EG / 64, 128, 0, stream>>>(src, tgt, x, gcoords, WT_ED1, ed_b1,
                                      WT_ED2, ed_b2, WT_MS1, ms_b1, WT_MS2,
                                      ms_b2, NF16, GF, CNT);
  k_upd<<<GG / 64, 128, 0, stream>>>(GF, CNT, WT_UP1, up_b1, WT_UP2, up_b2, T0);

  // ---- CNN ----
  k_stats<<<64, 128, 0, stream>>>(T0, ST, GG);
  k_bn_apply<<<(GG * HH + 255) / 256, 256, 0, stream>>>(
      T0, T0, ST, cnn_g, cnn_b, GG * HH, HH, 1.0f / GG, 0);

  run_block<128, 64>(T0, BUF1, BUF2, BUF1, BP[0], W1T[0], W2T[0], WST[0], ST + 512 + 0 * 1536, 9, 1, stream);
  run_block<64, 64>(BUF1, BUF2, BUF3, BUF2, BP[1], W1T[1], W2T[1], WST[1], ST + 512 + 1 * 1536, 9, 1, stream);
  run_block<64, 128>(BUF2, BUF1, BUF3, BUF1, BP[2], W1T[2], W2T[2], WST[2], ST + 512 + 2 * 1536, 9, 2, stream);
  run_block<128, 128>(BUF1, BUF2, BUF3, BUF2, BP[3], W1T[3], W2T[3], WST[3], ST + 512 + 3 * 1536, 5, 1, stream);
  run_block<128, 256>(BUF2, BUF1, BUF3, BUF1, BP[4], W1T[4], W2T[4], WST[4], ST + 512 + 4 * 1536, 5, 2, stream);
  run_block<256, 256>(BUF1, BUF2, BUF3, BUF2, BP[5], W1T[5], W2T[5], WST[5], ST + 512 + 5 * 1536, 3, 1, stream);

  k_pool_fc<<<BB, 256, 0, stream>>>(BUF2, fc_w, fc_b, (float*)d_out);
}
